// RNNLayer_4054449127655
// MI455X (gfx1250) — matrix operand order repaired and checked by reading, not compiled, not run on hardware
//
#include <hip/hip_runtime.h>
#include <hip/hip_bf16.h>

// Problem constants (B=64, T=512, I=512, H=1024)
#define B_   64
#define T_   512
#define I_   512
#define H_   1024
#define G3H_ 3072
#define BT_  (B_ * T_)

typedef __bf16 bf16_t;
typedef __attribute__((ext_vector_type(16))) __bf16 v16bf;
typedef __attribute__((ext_vector_type(8)))  __bf16 v8bf;
typedef __attribute__((ext_vector_type(8)))  float  v8f;

// ---------------------------------------------------------------------------
// CDNA5 async copy (global -> LDS), tracked with ASYNCcnt.
// ---------------------------------------------------------------------------
__device__ __forceinline__ void async_copy_b128(unsigned int lds_off, const bf16_t* g) {
    asm volatile("global_load_async_to_lds_b128 %0, %1, off"
                 :: "v"(lds_off), "v"(g) : "memory");
}
__device__ __forceinline__ void async_copy_b64(unsigned int lds_off, const bf16_t* g) {
    asm volatile("global_load_async_to_lds_b64 %0, %1, off"
                 :: "v"(lds_off), "v"(g) : "memory");
}
#define WAIT_ASYNC_0() asm volatile("s_wait_asynccnt 0x0" ::: "memory")
#define WAIT_ASYNC_2() asm volatile("s_wait_asynccnt 0x2" ::: "memory")
#define WAIT_ASYNC_3() asm volatile("s_wait_asynccnt 0x3" ::: "memory")

__device__ __forceinline__ unsigned int lds_addr_of(const void* p) {
    return (unsigned int)(uintptr_t)p;   // low 32 bits of generic ptr = LDS offset
}

// ---------------------------------------------------------------------------
// WMMA fragment helpers (CDNA5 wave32, V_WMMA_F32_16X16X32_BF16)
// A (16x32 bf16, row-major, lda elems): lane L: row=L&15, hi=L>>4;
//   elems 0..7 -> K = kbase+hi*8+j ; elems 8..15 -> K = kbase+hi*8+16+(j-8)
// B (32x16 bf16) packed tile-lane-major (512 bf16/tile, 32B per lane).
// ---------------------------------------------------------------------------
__device__ __forceinline__ v16bf load_a_frag(const bf16_t* A, int lda, int row0, int kbase) {
    int lane = threadIdx.x & 31;
    int r  = lane & 15;
    int hi = lane >> 4;
    const bf16_t* p = A + (size_t)(row0 + r) * lda + kbase + hi * 8;
    v8bf lo = *(const v8bf*)(p);
    v8bf hv = *(const v8bf*)(p + 16);
    v16bf out;
#pragma unroll
    for (int j = 0; j < 8; ++j) { out[j] = lo[j]; out[8 + j] = hv[j]; }
    return out;
}

__device__ __forceinline__ v16bf lds_b_frag(const bf16_t* sb) {
    int lane = threadIdx.x & 31;
    return *(const v16bf*)(sb + lane * 16);   // ds_load_b128 x2
}

__device__ __forceinline__ v8f wmma_bf16(v16bf a, v16bf b, v8f c) {
    return __builtin_amdgcn_wmma_f32_16x16x32_bf16(false, a, false, b, (short)0, c,
                                                   false, false);
}

__device__ __forceinline__ float fast_sigmoid(float x) {
    return 1.0f / (1.0f + __expf(-x));
}

// ---------------------------------------------------------------------------
// Kernel 1: LayerNorm over last dim (I=512) + convert to bf16. One wave/row.
// ---------------------------------------------------------------------------
__global__ void ln_bf16_kernel(const float* __restrict__ x, bf16_t* __restrict__ xn) {
    int wave = threadIdx.x >> 5;
    int lane = threadIdx.x & 31;
    int row  = blockIdx.x * 4 + wave;     // grid = BT/4, block = 128
    const float* xr = x + (size_t)row * I_;

    float v[16];
    float s = 0.0f;
#pragma unroll
    for (int j = 0; j < 16; ++j) { v[j] = xr[lane + j * 32]; s += v[j]; }
#pragma unroll
    for (int m = 16; m >= 1; m >>= 1) s += __shfl_xor(s, m, 32);
    float mu = s * (1.0f / (float)I_);

    float q = 0.0f;
#pragma unroll
    for (int j = 0; j < 16; ++j) { float d = v[j] - mu; q += d * d; }
#pragma unroll
    for (int m = 16; m >= 1; m >>= 1) q += __shfl_xor(q, m, 32);
    float rstd = __frsqrt_rn(q * (1.0f / (float)I_) + 1e-5f);

    bf16_t* orow = xn + (size_t)row * I_;
#pragma unroll
    for (int j = 0; j < 16; ++j) orow[lane + j * 32] = (bf16_t)((v[j] - mu) * rstd);
}

// ---------------------------------------------------------------------------
// Kernel 2: pack weight W [N rows, K cols] (row-major f32) into bf16 WMMA
// B-fragment tile order for W^T (K x N): tiles of 32x16, 512 elems, lane-major.
// ---------------------------------------------------------------------------
__global__ void pack_b_kernel(const float* __restrict__ W, bf16_t* __restrict__ out,
                              int N, int K) {
    int e = blockIdx.x * 256 + threadIdx.x;
    int total = N * K;
    if (e >= total) return;
    int tiles_n = N >> 4;
    int tile = e >> 9;
    int idx  = e & 511;
    int lane = idx >> 4;
    int j    = idx & 15;
    int hi   = lane >> 4;
    int cl   = lane & 15;
    int kt   = tile / tiles_n;
    int nt   = tile - kt * tiles_n;
    int k    = kt * 32 + hi * 8 + (j & 7) + ((j >> 3) << 4);
    int n    = nt * 16 + cl;
    out[e] = (bf16_t)W[(size_t)n * K + k];
}

// ---------------------------------------------------------------------------
// Kernel 3: zero the initial hidden state (f32 + bf16 copies).
// ---------------------------------------------------------------------------
__global__ void zero_h_kernel(float* __restrict__ hf, bf16_t* __restrict__ hb) {
    int i = blockIdx.x * 256 + threadIdx.x;   // grid covers B*H
    hf[i] = 0.0f;
    hb[i] = (bf16_t)0.0f;
}

// ---------------------------------------------------------------------------
// Kernel 4: gi = xn @ w_ih^T + b_ih, written as [T, B, 3H] f32.
// grid = (BT/64, 3H/64), block = 128 (4 waves). All 4 waves share the same
// 4 B tiles -> double-buffered async staging through LDS. K unrolled x2 with
// explicit even/odd buffers; all B frags loaded before the WMMA burst.
// ---------------------------------------------------------------------------
__global__ void gemm_gi_kernel(const bf16_t* __restrict__ A,        // xn [BT, I]
                               const bf16_t* __restrict__ Bp,       // packed w_ih^T
                               const float*  __restrict__ bias,     // b_ih [3H]
                               float* __restrict__ gi) {            // [T, B, 3H]
    __shared__ bf16_t sB[2][4 * 512];        // 4 tiles * 1KiB, double buffered
    int tid  = threadIdx.x;
    int wave = tid >> 5;
    int lane = tid & 31;
    int row0 = (blockIdx.x * 4 + wave) * 16;
    int nt0  = blockIdx.y * 4;
    const int tiles_n = G3H_ / 16;           // 192
    const int KT = I_ / 32;                  // 16 (even)

    auto issue = [&](int kt, int buf) {      // 2 x b128 per thread, 4KiB total
        const bf16_t* g = Bp + (size_t)(kt * tiles_n + nt0) * 512;
        unsigned int l = lds_addr_of(&sB[buf][0]) + tid * 16;
        async_copy_b128(l,        g + tid * 8);
        async_copy_b128(l + 2048, g + 1024 + tid * 8);
    };
    issue(0, 0);

    v8f acc[4];
#pragma unroll
    for (int i = 0; i < 4; ++i) acc[i] = (v8f)(0.0f);

    auto step = [&](int kt, int buf) {
        if (kt + 1 < KT) { issue(kt + 1, buf ^ 1); WAIT_ASYNC_2(); }
        else             { WAIT_ASYNC_0(); }
        __syncthreads();                      // round kt staged in all waves' view
        v16bf a = load_a_frag(A, I_, row0, kt * 32);
        const bf16_t* sb = &sB[buf][0];
        v16bf b0 = lds_b_frag(sb);            // clause of ds loads, single wait
        v16bf b1 = lds_b_frag(sb + 512);
        v16bf b2 = lds_b_frag(sb + 1024);
        v16bf b3 = lds_b_frag(sb + 1536);
        acc[0] = wmma_bf16(a, b0, acc[0]);    // back-to-back WMMA burst
        acc[1] = wmma_bf16(a, b1, acc[1]);
        acc[2] = wmma_bf16(a, b2, acc[2]);
        acc[3] = wmma_bf16(a, b3, acc[3]);
        __syncthreads();                      // buffer reusable for round kt+2
    };
    for (int kt = 0; kt < KT; kt += 2) {      // explicit even/odd buffers
        step(kt, 0);
        step(kt + 1, 1);
    }

    int hi = lane >> 4;
    int cl = lane & 15;
#pragma unroll
    for (int i = 0; i < 4; ++i) {
        int col = (nt0 + i) * 16 + cl;
        float bv = bias[col];
#pragma unroll
        for (int v = 0; v < 8; ++v) {
            int m = row0 + hi * 8 + v;          // global (b*T + t)
            int b = m >> 9;                      // /T_
            int t = m & (T_ - 1);
            gi[(size_t)(t * B_ + b) * G3H_ + col] = acc[i][v] + bv;
        }
    }
}

// ---------------------------------------------------------------------------
// Kernel 5: one GRU timestep, fully fused.
//   gh = h_prev @ w_hh^T + b_hh (3 gate tiles per column block, WMMA),
//   then r/z/n gate math, h_new = (1-z)*n + z*h.
// grid = H/16 = 64 blocks, block = 128 (wave w -> batch rows 16w..).
// ---------------------------------------------------------------------------
__global__ void gru_step_kernel(const float*  __restrict__ h_prev_f,   // [B, H]
                                const bf16_t* __restrict__ h_prev_b,   // [B, H]
                                const bf16_t* __restrict__ Wp,         // packed w_hh^T
                                const float*  __restrict__ b_hh,       // [3H]
                                const float*  __restrict__ gi_t,       // [B, 3H]
                                float*  __restrict__ h_new_f,          // [B, H]
                                bf16_t* __restrict__ h_new_b,          // [B, H]
                                bf16_t* __restrict__ h_hist,           // [BT, H]
                                int t) {
    __shared__ bf16_t sB[2][3 * 512];        // 3 gate tiles * 1KiB, double buffered
    int tid  = threadIdx.x;
    int wave = tid >> 5;
    int lane = tid & 31;
    int row0 = wave * 16;                    // batch rows
    int cb   = blockIdx.x;                   // 16-column block within H
    const int tiles_n = G3H_ / 16;           // 192
    const int KT = H_ / 32;                  // 32 (even)

    auto issue = [&](int kt, int buf) {      // one b64 per thread per gate tile
        unsigned int l = lds_addr_of(&sB[buf][0]) + tid * 8;
#pragma unroll
        for (int g = 0; g < 3; ++g) {
            const bf16_t* gp = Wp + (size_t)(kt * tiles_n + g * (H_ / 16) + cb) * 512;
            async_copy_b64(l + g * 1024, gp + tid * 4);
        }
    };
    issue(0, 0);

    v8f accR = (v8f)(0.0f), accZ = (v8f)(0.0f), accN = (v8f)(0.0f);

    auto step = [&](int kt, int buf) {
        if (kt + 1 < KT) { issue(kt + 1, buf ^ 1); WAIT_ASYNC_3(); }
        else             { WAIT_ASYNC_0(); }
        __syncthreads();
        v16bf a = load_a_frag(h_prev_b, H_, row0, kt * 32);
        const bf16_t* sb = &sB[buf][0];
        v16bf br = lds_b_frag(sb);
        v16bf bz = lds_b_frag(sb + 512);
        v16bf bn = lds_b_frag(sb + 1024);
        accR = wmma_bf16(a, br, accR);
        accZ = wmma_bf16(a, bz, accZ);
        accN = wmma_bf16(a, bn, accN);
        __syncthreads();
    };
    for (int kt = 0; kt < KT; kt += 2) {
        step(kt, 0);
        step(kt + 1, 1);
    }

    int hi  = lane >> 4;
    int cl  = lane & 15;
    int col = cb * 16 + cl;
    float bhr = b_hh[col];
    float bhz = b_hh[H_ + col];
    float bhn = b_hh[2 * H_ + col];

#pragma unroll
    for (int v = 0; v < 8; ++v) {
        int b = row0 + hi * 8 + v;                    // batch index
        const float* girow = gi_t + (size_t)b * G3H_;
        float ghr = accR[v] + bhr;
        float ghz = accZ[v] + bhz;
        float ghn = accN[v] + bhn;
        float r = fast_sigmoid(girow[col] + ghr);
        float z = fast_sigmoid(girow[H_ + col] + ghz);
        float n = tanhf(girow[2 * H_ + col] + r * ghn);
        float hp = h_prev_f[(size_t)b * H_ + col];
        float h  = (1.0f - z) * n + z * hp;
        h_new_f[(size_t)b * H_ + col] = h;
        bf16_t hb = (bf16_t)h;
        h_new_b[(size_t)b * H_ + col] = hb;
        h_hist[((size_t)b * T_ + t) * H_ + col] = hb;  // row b*T+t matches output GEMM
    }
}

// ---------------------------------------------------------------------------
// Kernel 6: out = x + h_hist @ w_out^T + b_out   ([BT, I] f32)
// grid = (BT/64, I/64), block = 128; same async-LDS B staging.
// ---------------------------------------------------------------------------
__global__ void gemm_out_kernel(const bf16_t* __restrict__ A,    // h_hist [BT, H]
                                const bf16_t* __restrict__ Bp,   // packed w_out^T
                                const float*  __restrict__ bias, // b_out [I]
                                const float*  __restrict__ x,    // [BT, I]
                                float* __restrict__ out) {       // [BT, I]
    __shared__ bf16_t sB[2][4 * 512];
    int tid  = threadIdx.x;
    int wave = tid >> 5;
    int lane = tid & 31;
    int row0 = (blockIdx.x * 4 + wave) * 16;
    int nt0  = blockIdx.y * 4;
    const int tiles_n = I_ / 16;             // 32
    const int KT = H_ / 32;                  // 32 (even)

    auto issue = [&](int kt, int buf) {
        const bf16_t* g = Bp + (size_t)(kt * tiles_n + nt0) * 512;
        unsigned int l = lds_addr_of(&sB[buf][0]) + tid * 16;
        async_copy_b128(l,        g + tid * 8);
        async_copy_b128(l + 2048, g + 1024 + tid * 8);
    };
    issue(0, 0);

    v8f acc[4];
#pragma unroll
    for (int i = 0; i < 4; ++i) acc[i] = (v8f)(0.0f);

    auto step = [&](int kt, int buf) {
        if (kt + 1 < KT) { issue(kt + 1, buf ^ 1); WAIT_ASYNC_2(); }
        else             { WAIT_ASYNC_0(); }
        __syncthreads();
        v16bf a = load_a_frag(A, H_, row0, kt * 32);
        const bf16_t* sb = &sB[buf][0];
        v16bf b0 = lds_b_frag(sb);
        v16bf b1 = lds_b_frag(sb + 512);
        v16bf b2 = lds_b_frag(sb + 1024);
        v16bf b3 = lds_b_frag(sb + 1536);
        acc[0] = wmma_bf16(a, b0, acc[0]);
        acc[1] = wmma_bf16(a, b1, acc[1]);
        acc[2] = wmma_bf16(a, b2, acc[2]);
        acc[3] = wmma_bf16(a, b3, acc[3]);
        __syncthreads();
    };
    for (int kt = 0; kt < KT; kt += 2) {
        step(kt, 0);
        step(kt + 1, 1);
    }

    int hi = lane >> 4;
    int cl = lane & 15;
#pragma unroll
    for (int i = 0; i < 4; ++i) {
        int col = (nt0 + i) * 16 + cl;
        float bv = bias[col];
#pragma unroll
        for (int v = 0; v < 8; ++v) {
            int m = row0 + hi * 8 + v;
            size_t idx = (size_t)m * I_ + col;
            out[idx] = x[idx] + acc[i][v] + bv;
        }
    }
}

// ---------------------------------------------------------------------------
// Host launcher. Workspace layout (256B-aligned slabs):
//   xn_bf16 [BT,I] 32MiB | wih_pk 3MiB | whh_pk 6MiB | wout_pk 1MiB |
//   gi [T,B,3H] f32 384MiB | h_hist [BT,H] bf16 64MiB | h ping-pong buffers
// ---------------------------------------------------------------------------
extern "C" void kernel_launch(void* const* d_in, const int* in_sizes, int n_in,
                              void* d_out, int out_size, void* d_ws, size_t ws_size,
                              hipStream_t stream) {
    const float* x     = (const float*)d_in[0];
    const float* w_ih  = (const float*)d_in[1];
    const float* w_hh  = (const float*)d_in[2];
    const float* b_ih  = (const float*)d_in[3];
    const float* b_hh  = (const float*)d_in[4];
    const float* w_out = (const float*)d_in[5];
    const float* b_out = (const float*)d_in[6];
    float* out = (float*)d_out;

    char* ws = (char*)d_ws;
    size_t off = 0;
    auto alloc = [&](size_t bytes) {
        char* p = ws + off;
        off += (bytes + 255) & ~(size_t)255;
        return p;
    };
    bf16_t* xn      = (bf16_t*)alloc((size_t)BT_ * I_ * 2);
    bf16_t* wih_pk  = (bf16_t*)alloc((size_t)I_ * G3H_ * 2);
    bf16_t* whh_pk  = (bf16_t*)alloc((size_t)H_ * G3H_ * 2);
    bf16_t* wout_pk = (bf16_t*)alloc((size_t)H_ * I_ * 2);
    float*  gi      = (float*) alloc((size_t)BT_ * G3H_ * 4);
    bf16_t* h_hist  = (bf16_t*)alloc((size_t)BT_ * H_ * 2);
    float*  hA_f    = (float*) alloc((size_t)B_ * H_ * 4);
    float*  hB_f    = (float*) alloc((size_t)B_ * H_ * 4);
    bf16_t* hA_b    = (bf16_t*)alloc((size_t)B_ * H_ * 2);
    bf16_t* hB_b    = (bf16_t*)alloc((size_t)B_ * H_ * 2);

    // 1) LayerNorm + bf16 convert
    ln_bf16_kernel<<<BT_ / 4, 128, 0, stream>>>(x, xn);

    // 2) Pack weights into WMMA B-fragment tile order
    pack_b_kernel<<<(G3H_ * I_ + 255) / 256, 256, 0, stream>>>(w_ih, wih_pk, G3H_, I_);
    pack_b_kernel<<<(G3H_ * H_ + 255) / 256, 256, 0, stream>>>(w_hh, whh_pk, G3H_, H_);
    pack_b_kernel<<<(I_ * H_ + 255) / 256, 256, 0, stream>>>(w_out, wout_pk, I_, H_);

    // 3) h0 = 0
    zero_h_kernel<<<(B_ * H_) / 256, 256, 0, stream>>>(hA_f, hA_b);

    // 4) gi = xn @ w_ih^T + b_ih   -> [T, B, 3H]
    gemm_gi_kernel<<<dim3(BT_ / 64, G3H_ / 64), 128, 0, stream>>>(xn, wih_pk, b_ih, gi);

    // 5) Sequential GRU: one fused kernel per timestep (ping-pong h buffers)
    float*  hf[2] = { hA_f, hB_f };
    bf16_t* hb[2] = { hA_b, hB_b };
    for (int t = 0; t < T_; ++t) {
        int s = t & 1, d = (t + 1) & 1;
        gru_step_kernel<<<H_ / 16, 128, 0, stream>>>(
            hf[s], hb[s], whh_pk, b_hh,
            gi + (size_t)t * B_ * G3H_,
            hf[d], hb[d], h_hist, t);
    }

    // 6) out = x + h_hist @ w_out^T + b_out
    gemm_out_kernel<<<dim3(BT_ / 64, I_ / 64), 128, 0, stream>>>(
        h_hist, wout_pk, b_out, x, out);

    (void)in_sizes; (void)n_in; (void)out_size; (void)ws_size;
}
